// VGG_83794811945245
// MI455X (gfx1250) — compile-verified, matrix operand order checked by reading, not run on hardware
//
#include <hip/hip_runtime.h>
#include <hip/hip_bf16.h>
#include <math.h>

// ---------------------------------------------------------------------------
// AdderNet-VGG on MI455X (gfx1250):
//   * 8 fused adder-conv(+ReLU+BN) layers: VALU fp32; x-planes staged in LDS
//     via b128 copies, weight slices staged in LDS (broadcast reads),
//     global_prefetch for the next Cin chunk.
//   * final FC via v_wmma_f32_16x16x32_f16 with f16 operand tiles in LDS
//     (fully unrolled K loop -> 16 static WMMA, branchless inner loop).
// ---------------------------------------------------------------------------

typedef __attribute__((ext_vector_type(16))) _Float16 v16h;
typedef __attribute__((ext_vector_type(8)))  _Float16 v8h;
typedef __attribute__((ext_vector_type(8)))  float    v8f;

// ---------------------------------------------------------------------------
// Fused adder-conv + ReLU + BN.
//   x:   [B, Cin, H, W]     w: [Cout, Cin, 3, 3]    bn: [4, Cout]
//   out: [B, Cout, OH, OW]
// Block = (OH*OW, CT); grid = (1, Cout/CT, B).
// Per Cin-chunk: stage CHUNK x-planes (float4 -> ds b128) and, if WLDS,
// the CT*CHUNK*9 weight slice into LDS; accumulate acc -= |x - w|.
// ---------------------------------------------------------------------------
template<int Cin, int H, int W, int OH, int OW, int S, int CT, int CHUNK, bool WLDS>
__global__ __launch_bounds__(512)
void adder_block(const float* __restrict__ x,
                 const float* __restrict__ w,
                 const float* __restrict__ bn,
                 float* __restrict__ out, int Cout) {
    __shared__ __align__(16) float lx[CHUNK * H * W];
    __shared__ float lw[WLDS ? (CT * CHUNK * 9) : 1];

    const int b    = blockIdx.z;
    const int ty   = threadIdx.y;
    const int cout = blockIdx.y * CT + ty;
    const int sp   = threadIdx.x;              // < OH*OW
    const int oh   = sp / OW;
    const int ow   = sp % OW;
    const int tid  = ty * (OH * OW) + threadIdx.x;
    const int nthr = CT * OH * OW;

    const float* __restrict__ xb = x + (size_t)b * Cin * H * W;
    // first cout handled by this block (for cooperative weight staging)
    const float* __restrict__ wblk = w + (size_t)(blockIdx.y * CT) * Cin * 9;

    float acc = 0.0f;

    for (int c0 = 0; c0 < Cin; c0 += CHUNK) {
        __syncthreads();
        // --- stage x planes: global_load_b128 -> ds_store_b128 ---
        {
            const float4* __restrict__ src = (const float4*)(xb + (size_t)c0 * H * W);
            float4* __restrict__ dst = (float4*)lx;
            constexpr int N4 = (CHUNK * H * W) / 4;
            for (int idx = tid; idx < N4; idx += nthr)
                dst[idx] = src[idx];
        }
        // --- stage weight slice: lw[ty][c][tap], contiguous per cout ---
        if (WLDS) {
            constexpr int NW = CT * CHUNK * 9;      // elements to stage
            constexpr int WR = CHUNK * 9;           // per-cout row length
            for (int idx = tid; idx < NW; idx += nthr) {
                const int tyw = idx / WR;
                const int rem = idx - tyw * WR;
                lw[idx] = wblk[(size_t)tyw * Cin * 9 + (size_t)c0 * 9 + rem];
            }
        }
        // --- prefetch next x chunk (global_prefetch_b8) ---
        if (c0 + CHUNK < Cin) {
            const int poff = tid * 16;
            if (poff < CHUNK * H * W)
                __builtin_prefetch(xb + (size_t)(c0 + CHUNK) * H * W + poff, 0, 1);
        }
        __syncthreads();

        const float* __restrict__ wbase =
            WLDS ? (lw + ty * (CHUNK * 9))
                 : (w + ((size_t)cout * Cin + c0) * 9);

        #pragma unroll 4
        for (int c = 0; c < CHUNK; ++c) {
            const float* __restrict__ xp = lx + c * H * W + (oh * S) * W + (ow * S);
            const float* __restrict__ wc = wbase + c * 9;
            #pragma unroll
            for (int i = 0; i < 3; ++i) {
                #pragma unroll
                for (int j = 0; j < 3; ++j) {
                    acc -= fabsf(xp[i * W + j] - wc[i * 3 + j]);
                }
            }
        }
    }

    // ReLU then BN (reference: bn(relu(adder)))
    const float g  = bn[cout];
    const float be = bn[Cout + cout];
    const float m  = bn[2 * Cout + cout];
    const float v  = bn[3 * Cout + cout];
    const float inv = g / sqrtf(v + 1e-5f);
    const float r   = fmaxf(acc, 0.0f);
    out[((size_t)b * Cout + cout) * (OH * OW) + sp] = r * inv + (be - m * inv);
}

// ---------------------------------------------------------------------------
// FC: out[32,10] = h[32,512] @ fc_w[10,512]^T + fc_b   via WMMA f16->f32.
// One wave per 16-row M tile. A tile (16x512) and zero-padded B tile (16x512,
// rows 10..15 = 0) are converted to f16 and staged in LDS once; the K loop is
// branchless: 4x ds_load_b128 + v_wmma per 32-wide K step, fully unrolled.
// Fragment layout (ISA 7.12.2, 16-bit 16x32): lanes 0-15 hold K base k0,
// lanes 16-31 hold k0+8; components 0..7 -> K+0..7, components 8..15 -> K+16..23.
// C/D: lane%16 = N, VGPR r = M row r (+8 for lanes 16-31).
// ---------------------------------------------------------------------------
__global__ __launch_bounds__(32)
void fc_wmma(const float* __restrict__ h,     // [32, 512]
             const float* __restrict__ fcw,   // [10, 512]
             const float* __restrict__ fcb,   // [10]
             float* __restrict__ out) {       // [32, 10]
    __shared__ __align__(16) _Float16 lA[16 * 512];   // 16 KB
    __shared__ __align__(16) _Float16 lB[16 * 512];   // 16 KB

    const int lane  = threadIdx.x;            // 0..31
    const int mtile = blockIdx.x;             // 0..1
    const int lrow  = lane & 15;              // M row (A frag) / N col (B frag)
    const int hi16  = (lane >= 16) ? 8 : 0;

    // --- stage tiles: f32 global b128 loads -> v_cvt_pk_f16_f32 -> LDS ---
    for (int i = lane * 4; i < 16 * 512; i += 32 * 4) {
        const int row = i >> 9;
        const int k   = i & 511;
        const float4 va = *(const float4*)(h + (size_t)(mtile * 16 + row) * 512 + k);
        lA[i + 0] = (_Float16)va.x; lA[i + 1] = (_Float16)va.y;
        lA[i + 2] = (_Float16)va.z; lA[i + 3] = (_Float16)va.w;
        float4 vb = make_float4(0.f, 0.f, 0.f, 0.f);
        if (row < 10)
            vb = *(const float4*)(fcw + (size_t)row * 512 + k);
        lB[i + 0] = (_Float16)vb.x; lB[i + 1] = (_Float16)vb.y;
        lB[i + 2] = (_Float16)vb.z; lB[i + 3] = (_Float16)vb.w;
    }
    __syncthreads();

    const _Float16* __restrict__ arow = lA + lrow * 512 + hi16;
    const _Float16* __restrict__ brow = lB + lrow * 512 + hi16;

    v8f acc = {};
    #pragma unroll
    for (int k0 = 0; k0 < 512; k0 += 32) {
        const v8h a_lo = *(const v8h*)(arow + k0);        // ds_load_b128
        const v8h a_hi = *(const v8h*)(arow + k0 + 16);
        const v8h b_lo = *(const v8h*)(brow + k0);
        const v8h b_hi = *(const v8h*)(brow + k0 + 16);
        v16h a, bfrag;
        #pragma unroll
        for (int c = 0; c < 8; ++c) {
            a[c] = a_lo[c];     a[c + 8] = a_hi[c];
            bfrag[c] = b_lo[c]; bfrag[c + 8] = b_hi[c];
        }
        acc = __builtin_amdgcn_wmma_f32_16x16x32_f16(
            /*neg_a=*/false, a, /*neg_b=*/false, bfrag,
            /*c_mod=*/(short)0, acc, /*reuse_a=*/false, /*reuse_b=*/false);
    }

    const int n = lane & 15;
    if (n < 10) {
        const float bias = fcb[n];
        #pragma unroll
        for (int r = 0; r < 8; ++r) {
            const int m = mtile * 16 + r + hi16;
            out[m * 10 + n] = acc[r] + bias;
        }
    }
}

// ---------------------------------------------------------------------------
// Launch: dict-order inputs: x, w1, bn1, w2, bn2, ..., w8, bn8, fc_w, fc_b.
// Activations ping-pong between two 4MB regions of d_ws.
// ---------------------------------------------------------------------------
extern "C" void kernel_launch(void* const* d_in, const int* in_sizes, int n_in,
                              void* d_out, int out_size, void* d_ws, size_t ws_size,
                              hipStream_t stream) {
    (void)in_sizes; (void)n_in; (void)out_size; (void)ws_size;

    const float* x = (const float*)d_in[0];
    const float* w[8];
    const float* bn[8];
    for (int l = 0; l < 8; ++l) {
        w[l]  = (const float*)d_in[1 + 2 * l];
        bn[l] = (const float*)d_in[2 + 2 * l];
    }
    const float* fcw = (const float*)d_in[17];
    const float* fcb = (const float*)d_in[18];

    float* bufA = (float*)d_ws;                 // 4MB region (max layer = 3.97MB)
    float* bufB = bufA + (size_t)(1u << 20);    // second 4MB region

    // L1: 3->64, 32x32 -> 15x15, stride 2
    adder_block<3, 32, 32, 15, 15, 2, 2, 3, true>
        <<<dim3(1, 32, 32), dim3(225, 2), 0, stream>>>(x, w[0], bn[0], bufA, 64);
    // L2: 64->128, 15x15 -> 13x13
    adder_block<64, 15, 15, 13, 13, 1, 2, 32, true>
        <<<dim3(1, 64, 32), dim3(169, 2), 0, stream>>>(bufA, w[1], bn[1], bufB, 128);
    // L3: 128->256, 13x13 -> 11x11
    adder_block<128, 13, 13, 11, 11, 1, 4, 32, true>
        <<<dim3(1, 64, 32), dim3(121, 4), 0, stream>>>(bufB, w[2], bn[2], bufA, 256);
    // L4: 256->256, 11x11 -> 9x9
    adder_block<256, 11, 11, 9, 9, 1, 4, 32, true>
        <<<dim3(1, 64, 32), dim3(81, 4), 0, stream>>>(bufA, w[3], bn[3], bufB, 256);
    // L5: 256->512, 9x9 -> 7x7
    adder_block<256, 9, 9, 7, 7, 1, 8, 32, true>
        <<<dim3(1, 64, 32), dim3(49, 8), 0, stream>>>(bufB, w[4], bn[4], bufA, 512);
    // L6: 512->512, 7x7 -> 5x5
    adder_block<512, 7, 7, 5, 5, 1, 16, 32, true>
        <<<dim3(1, 32, 32), dim3(25, 16), 0, stream>>>(bufA, w[5], bn[5], bufB, 512);
    // L7: 512->512, 5x5 -> 3x3
    adder_block<512, 5, 5, 3, 3, 1, 32, 32, true>
        <<<dim3(1, 16, 32), dim3(9, 32), 0, stream>>>(bufB, w[6], bn[6], bufA, 512);
    // L8: 512->512, 3x3 -> 1x1  (no spatial weight reuse -> direct global weights)
    adder_block<512, 3, 3, 1, 1, 1, 256, 32, false>
        <<<dim3(1, 2, 32), dim3(1, 256), 0, stream>>>(bufA, w[7], bn[7], bufB, 512);

    // FC: [32,512] @ [512,10] + b  -> d_out (f32), via WMMA
    fc_wmma<<<dim3(2), dim3(32), 0, stream>>>(bufB, fcw, fcb, (float*)d_out);
}
